// RowSelfAttention_12472585027590
// MI455X (gfx1250) — compile-verified, matrix operand order checked by reading, not run on hardware
//
#include <hip/hip_runtime.h>
#include <hip/hip_bf16.h>

// ---------------------------------------------------------------------------
// RowSelfAttention for MI455X (gfx1250): bf16 WMMA pipeline, f32 accumulate,
// async global->LDS copies (ASYNCcnt) + TDM tensor_load_to_lds (TENSORcnt).
// Shapes: R=C=256, B=1, E=768, H=12, Dh=64.
// ---------------------------------------------------------------------------

typedef __attribute__((ext_vector_type(16))) __bf16 v16bf;
typedef __attribute__((ext_vector_type(8)))  float  v8f;
typedef unsigned int u32x4 __attribute__((ext_vector_type(4)));
typedef int          i32x4 __attribute__((ext_vector_type(4)));
typedef int          i32x8 __attribute__((ext_vector_type(8)));
typedef int          v4i   __attribute__((ext_vector_type(4)));

typedef __attribute__((address_space(1))) v4i g_v4i;   // global 16B chunk
typedef __attribute__((address_space(3))) v4i l_v4i;   // LDS 16B chunk
typedef __attribute__((address_space(3))) void lds_void;

union FragAB { v16bf v; unsigned int u[8]; };
union FragC  { v8f  v; float f[8]; };

// ---- CDNA5 feature probes (compile-only safety: fall back to sync paths) ---
#if defined(__HIP_DEVICE_COMPILE__) && __has_builtin(__builtin_amdgcn_global_load_async_to_lds_b128)
#define HAVE_ASYNC_LDS 1
#else
#define HAVE_ASYNC_LDS 0
#endif

#if defined(__HIP_DEVICE_COMPILE__) && __has_builtin(__builtin_amdgcn_tensor_load_to_lds)
#define HAVE_TDM 1
#else
#define HAVE_TDM 0
#endif

template <int N>
__device__ __forceinline__ void wait_asynccnt() {
#if HAVE_ASYNC_LDS
#if __has_builtin(__builtin_amdgcn_s_wait_asynccnt)
    __builtin_amdgcn_s_wait_asynccnt((short)N);
#else
    asm volatile("s_wait_asynccnt %0" ::"i"(N) : "memory");
#endif
#endif
}

__device__ __forceinline__ void wait_tensorcnt0() {
#if HAVE_TDM
#if __has_builtin(__builtin_amdgcn_s_wait_tensorcnt)
    __builtin_amdgcn_s_wait_tensorcnt((short)0);
#else
    asm volatile("s_wait_tensorcnt 0x0" ::: "memory");
#endif
#endif
}

// 16-byte global -> LDS copy (async when available)
__device__ __forceinline__ void copy16_g2lds(const unsigned short* __restrict__ g,
                                             unsigned short* l) {
#if HAVE_ASYNC_LDS
    __builtin_amdgcn_global_load_async_to_lds_b128((g_v4i*)g, (l_v4i*)l, 0, 0);
#else
    *(uint4*)l = *(const uint4*)g;
#endif
}

__device__ __forceinline__ unsigned short f32_to_bf16(float f) {
    unsigned int u = __builtin_bit_cast(unsigned int, f);
    unsigned int r = u + 0x7FFFu + ((u >> 16) & 1u);   // round-to-nearest-even
    return (unsigned short)(r >> 16);
}

__device__ __forceinline__ v8f wmma_bf16(const FragAB& a, const FragAB& b, const v8f& c) {
    return __builtin_amdgcn_wmma_f32_16x16x32_bf16(
        false, a.v, false, b.v, (short)0, c, false, false);
}

// Build a bf16 fragment from a 32-ushort LDS row (ISA 16-bit A/B layout:
// lanes 0-15 hold K in {kb..kb+7, kb+16..kb+23}, kb = (lane>>4)*8).
__device__ __forceinline__ void load_frag(FragAB& f, const unsigned short* rowPtr, int kb) {
    const unsigned int* p = (const unsigned int*)rowPtr;
    int o = kb >> 1;
#pragma unroll
    for (int i = 0; i < 4; ++i) { f.u[i] = p[o + i]; f.u[i + 4] = p[8 + o + i]; }
}

// ---------------------------------------------------------------------------
// 1) f32 -> bf16 elementwise convert (vectorized x4)
// ---------------------------------------------------------------------------
__global__ __launch_bounds__(256)
void convert_f32_bf16_kernel(const float* __restrict__ src,
                             unsigned short* __restrict__ dst, long long n4) {
    long long i = (long long)blockIdx.x * blockDim.x + threadIdx.x;
    if (i >= n4) return;
    float4 f = ((const float4*)src)[i];
    ushort4 o;
    o.x = f32_to_bf16(f.x); o.y = f32_to_bf16(f.y);
    o.z = f32_to_bf16(f.z); o.w = f32_to_bf16(f.w);
    ((ushort4*)dst)[i] = o;
}

// ---------------------------------------------------------------------------
// 2) Wt[k*768+n] = bf16(W[n*768+k])   (768x768, LDS transpose)
// ---------------------------------------------------------------------------
__global__ __launch_bounds__(256)
void transpose_convert_768_kernel(const float* __restrict__ W,
                                  unsigned short* __restrict__ Wt) {
    __shared__ float tile[32][33];
    int bx = blockIdx.x * 32, by = blockIdx.y * 32;
    int tx = threadIdx.x, ty = threadIdx.y;   // block (32, 8)
#pragma unroll
    for (int yy = ty; yy < 32; yy += 8)
        tile[yy][tx] = W[(size_t)(by + yy) * 768 + bx + tx];
    __syncthreads();
#pragma unroll
    for (int yy = ty; yy < 32; yy += 8)
        Wt[(size_t)(bx + yy) * 768 + by + tx] = f32_to_bf16(tile[tx][yy]);
}

// ---------------------------------------------------------------------------
// 3) Generic bf16 WMMA GEMM: C[M,N] = (A[M,K] * B[K,N] + bias) * scale
//    BM=128, BN=64, BK=32; 256 threads = 8 waves; wave tile 32x32 (2x2 WMMA).
//    A tile staged via TDM tensor_load_to_lds (wave 0 issues the descriptor).
// ---------------------------------------------------------------------------
#define GM_BM 128
#define GM_BN 64
#define GM_BK 32

template <bool OUT_BF16>
__global__ __launch_bounds__(256)
void gemm_bf16_kernel(const unsigned short* __restrict__ A,
                      const unsigned short* __restrict__ B,
                      void* __restrict__ Cout,
                      int M, int N, int K,
                      const float* __restrict__ bias, float scale) {
    __shared__ __align__(16) unsigned short sA[GM_BM][GM_BK];   // [m][k]
    __shared__ __align__(16) unsigned short sBt[GM_BN][GM_BK];  // [n][k]

    const int tid  = threadIdx.x;
    const int m0   = blockIdx.x * GM_BM;
    const int n0   = blockIdx.y * GM_BN;
    const int lane = tid & 31, wave = tid >> 5;
    const int wm   = (wave & 3) * 32;
    const int wn   = (wave >> 2) * 32;
    const int fr   = lane & 15;
    const int kb   = (lane >> 4) * 8;

    const int arow = tid >> 1;
    const int aks  = (tid & 1) * 16;
    const int bflat = tid * 8;
    const int bkr   = bflat >> 6;          // 0..31
    const int bnc   = bflat & 63;          // multiple of 8

    FragC acc[2][2];
#pragma unroll
    for (int i = 0; i < 2; ++i)
#pragma unroll
        for (int j = 0; j < 2; ++j) acc[i][j].v = {};

    for (int kt = 0; kt < K; kt += GM_BK) {
#if HAVE_TDM
        if (wave == 0) {
            // D# for a 2D bf16 tile: 32 (contig, stride 1) x 128 rows (stride K)
            unsigned long long ga =
                (unsigned long long)(const void*)(A + (size_t)m0 * K + kt);
            unsigned ldsb = (unsigned)(unsigned long long)(lds_void*)&sA[0][0];
            u32x4 g0 = { 1u,                                    // count=1, flags=0
                         ldsb,                                  // lds_addr
                         (unsigned)(ga & 0xFFFFFFFFull),
                         (unsigned)((ga >> 32) & 0x01FFFFFFull) | (2u << 30) }; // type=2
            i32x8 g1 = { (int)(1u << 16),          // data_size = 2 bytes
                         (int)((unsigned)GM_BK << 16),   // tensor_dim0 = 32
                         (int)((unsigned)GM_BM << 16),   // dim0_hi=0 | tensor_dim1 = 128
                         (int)((unsigned)GM_BK << 16),   // dim1_hi=0 | tile_dim0 = 32
                         (int)GM_BM,                     // tile_dim1 = 128, tile_dim2 = 0
                         (int)K,                         // tensor_dim0_stride lo
                         0, 0 };
            i32x4 gz = { 0, 0, 0, 0 };
            i32x8 gz8 = { 0, 0, 0, 0, 0, 0, 0, 0 };
            __builtin_amdgcn_tensor_load_to_lds(g0, g1, gz, gz, gz8, 0);
        }
#else
        copy16_g2lds(A + (size_t)(m0 + arow) * K + kt + aks,     &sA[arow][aks]);
        copy16_g2lds(A + (size_t)(m0 + arow) * K + kt + aks + 8, &sA[arow][aks + 8]);
#endif
        // B tile: load 8 bf16 and scatter transposed into LDS
        uint4 bd = *(const uint4*)(B + (size_t)(kt + bkr) * N + n0 + bnc);
        unsigned short btmp[8];
        *(uint4*)btmp = bd;
#pragma unroll
        for (int e = 0; e < 8; ++e) sBt[bnc + e][bkr] = btmp[e];

        if (kt + GM_BK < K)   // lowers to global_prefetch_b8
            __builtin_prefetch(B + (size_t)(kt + GM_BK + bkr) * N + n0 + bnc, 0, 1);

#if HAVE_TDM
        if (wave == 0) wait_tensorcnt0();
#endif
        wait_asynccnt<0>();
        __syncthreads();

        FragAB af[2], bf[2];
#pragma unroll
        for (int ms = 0; ms < 2; ++ms) load_frag(af[ms], &sA[wm + ms * 16 + fr][0], kb);
#pragma unroll
        for (int ns = 0; ns < 2; ++ns) load_frag(bf[ns], &sBt[wn + ns * 16 + fr][0], kb);
#pragma unroll
        for (int ms = 0; ms < 2; ++ms)
#pragma unroll
            for (int ns = 0; ns < 2; ++ns)
                acc[ms][ns].v = wmma_bf16(af[ms], bf[ns], acc[ms][ns].v);
        __syncthreads();
    }

    const int cn = lane & 15, rb = (lane >> 4) * 8;
#pragma unroll
    for (int ms = 0; ms < 2; ++ms)
#pragma unroll
        for (int ns = 0; ns < 2; ++ns) {
            int col = n0 + wn + ns * 16 + cn;
            float bv = bias ? bias[col] : 0.0f;
#pragma unroll
            for (int v = 0; v < 8; ++v) {
                int row = m0 + wm + ms * 16 + rb + v;
                float val = (acc[ms][ns].f[v] + bv) * scale;
                if (OUT_BF16)
                    ((unsigned short*)Cout)[(size_t)row * N + col] = f32_to_bf16(val);
                else
                    __builtin_nontemporal_store(val, (float*)Cout + (size_t)row * N + col);
            }
        }
}

// ---------------------------------------------------------------------------
// 4) Scores: s[h,i,j] = sum_{r,d} q[r,i,h,d] * k[r,j,h,d]
//    Per head 256x256, K = 16384. Block tile 64x64, BK=32.
//    Double-buffered async global->LDS pipeline over 512 K-steps.
// ---------------------------------------------------------------------------
__global__ __launch_bounds__(256)
void scores_kernel(const unsigned short* __restrict__ q,
                   const unsigned short* __restrict__ k,
                   float* __restrict__ scores) {  // [H][256][256]
    __shared__ __align__(16) unsigned short sQ[2][64][32];
    __shared__ __align__(16) unsigned short sK[2][64][32];

    const int tid = threadIdx.x;
    const int i0 = blockIdx.x * 64, j0 = blockIdx.y * 64, h = blockIdx.z;
    const int lane = tid & 31, wave = tid >> 5;
    const int wm = (wave & 3) * 16;
    const int wn = (wave >> 2) * 32;
    const int fr = lane & 15, kb = (lane >> 4) * 8;

    const int lrow = tid >> 2;           // 0..63
    const int lks  = (tid & 3) * 8;      // 0,8,16,24

    FragC acc[2];
    acc[0].v = {}; acc[1].v = {};

    auto issue_tile = [&](int it) {
        int r = it >> 1;
        int db = (it & 1) * 32;
        int b = it & 1;
        size_t rbase = (size_t)r * 256;
        copy16_g2lds(q + ((rbase + i0 + lrow) * 768) + h * 64 + db + lks,
                     &sQ[b][lrow][lks]);
        copy16_g2lds(k + ((rbase + j0 + lrow) * 768) + h * 64 + db + lks,
                     &sK[b][lrow][lks]);
    };

    issue_tile(0);
    for (int it = 0; it < 512; ++it) {
        if (it + 1 < 512) {
            issue_tile(it + 1);       // prefetch next tile into other buffer
            wait_asynccnt<2>();       // current buffer's 2 copies complete
        } else {
            wait_asynccnt<0>();
        }
        __syncthreads();

        int b = it & 1;
        FragAB af, bf0, bf1;
        load_frag(af,  &sQ[b][wm + fr][0], kb);
        load_frag(bf0, &sK[b][wn + fr][0], kb);
        load_frag(bf1, &sK[b][wn + 16 + fr][0], kb);
        acc[0].v = wmma_bf16(af, bf0, acc[0].v);
        acc[1].v = wmma_bf16(af, bf1, acc[1].v);
        __syncthreads();
    }

    const int cn = lane & 15, rb = (lane >> 4) * 8;
#pragma unroll
    for (int ns = 0; ns < 2; ++ns)
#pragma unroll
        for (int v = 0; v < 8; ++v) {
            int i = i0 + wm + rb + v;
            int j = j0 + wn + ns * 16 + cn;
            scores[((size_t)h * 256 + i) * 256 + j] = acc[ns].f[v];
        }
}

// ---------------------------------------------------------------------------
// 5) Softmax over j; writes f32 probs (d_out) + bf16 copy for context GEMM.
// ---------------------------------------------------------------------------
__global__ __launch_bounds__(256)
void softmax_kernel(float* __restrict__ probs,
                    unsigned short* __restrict__ probs_bf) {
    __shared__ float red[256];
    const int row = blockIdx.x;
    const int j = threadIdx.x;
    float x = probs[(size_t)row * 256 + j];
    red[j] = x;
    __syncthreads();
    for (int s = 128; s > 0; s >>= 1) {
        if (j < s) red[j] = fmaxf(red[j], red[j + s]);
        __syncthreads();
    }
    float m = red[0];
    __syncthreads();
    float e = __expf(x - m);
    red[j] = e;
    __syncthreads();
    for (int s = 128; s > 0; s >>= 1) {
        if (j < s) red[j] += red[j + s];
        __syncthreads();
    }
    float p = e * (1.0f / red[0]);
    probs[(size_t)row * 256 + j] = p;
    probs_bf[(size_t)row * 256 + j] = f32_to_bf16(p);
}

// ---------------------------------------------------------------------------
// 6) Context: ctx[r,i,h,d] = sum_j P[h,i,j] * v[r,j,h,d]
//    Per (r,h): 256x64 GEMM, K=256. Block tile 64(i) x 64(d), BK=32.
//    P tile staged with async global->LDS; V tile transposed via VGPRs.
// ---------------------------------------------------------------------------
__global__ __launch_bounds__(256)
void context_kernel(const unsigned short* __restrict__ pbf,  // [H][256][256]
                    const unsigned short* __restrict__ v,    // [R*C][768]
                    unsigned short* __restrict__ ctx) {      // [R*C][768]
    __shared__ __align__(16) unsigned short sP[64][32];   // [i][j]
    __shared__ __align__(16) unsigned short sVt[64][32];  // [d][j]

    const int tid = threadIdx.x;
    const int i0 = blockIdx.x * 64, h = blockIdx.y, r = blockIdx.z;
    const int lane = tid & 31, wave = tid >> 5;
    const int wm = (wave & 3) * 16, wn = (wave >> 2) * 32;
    const int fr = lane & 15, kb = (lane >> 4) * 8;

    const int prow = tid >> 2, pks = (tid & 3) * 8;     // P tile: 64x32
    const int vjr = tid >> 3, vd0 = (tid & 7) * 8;      // V tile: 32(j) x 64(d)

    FragC acc[2];
    acc[0].v = {}; acc[1].v = {};

    for (int jt = 0; jt < 256; jt += 32) {
        copy16_g2lds(pbf + ((size_t)h * 256 + i0 + prow) * 256 + jt + pks,
                     &sP[prow][pks]);

        size_t voff = ((size_t)(r * 256 + jt + vjr) * 768) + h * 64 + vd0;
        uint4 vd = *(const uint4*)(v + voff);
        unsigned short vtmp[8];
        *(uint4*)vtmp = vd;
#pragma unroll
        for (int e = 0; e < 8; ++e) sVt[vd0 + e][vjr] = vtmp[e];

        wait_asynccnt<0>();
        __syncthreads();

        FragAB af, bf0, bf1;
        load_frag(af,  &sP[wm + fr][0], kb);
        load_frag(bf0, &sVt[wn + fr][0], kb);
        load_frag(bf1, &sVt[wn + 16 + fr][0], kb);
        acc[0].v = wmma_bf16(af, bf0, acc[0].v);
        acc[1].v = wmma_bf16(af, bf1, acc[1].v);
        __syncthreads();
    }

    const int cn = lane & 15, rb = (lane >> 4) * 8;
#pragma unroll
    for (int ns = 0; ns < 2; ++ns)
#pragma unroll
        for (int v8 = 0; v8 < 8; ++v8) {
            int i = i0 + wm + rb + v8;
            int d = wn + ns * 16 + cn;
            ctx[((size_t)(r * 256 + i)) * 768 + h * 64 + d] = f32_to_bf16(acc[ns].f[v8]);
        }
}

// ---------------------------------------------------------------------------
// Host-side orchestration
// ---------------------------------------------------------------------------
extern "C" void kernel_launch(void* const* d_in, const int* in_sizes, int n_in,
                              void* d_out, int out_size, void* d_ws, size_t ws_size,
                              hipStream_t stream) {
    (void)in_sizes; (void)n_in; (void)out_size; (void)ws_size;
    const float* x  = (const float*)d_in[0];
    const float* Wq = (const float*)d_in[1];
    const float* bq = (const float*)d_in[2];
    const float* Wk = (const float*)d_in[3];
    const float* bk = (const float*)d_in[4];
    const float* Wv = (const float*)d_in[5];
    const float* bv = (const float*)d_in[6];
    const float* Wo = (const float*)d_in[7];
    const float* bo = (const float*)d_in[8];

    const size_t M = 65536;   // R*C*B
    const size_t E = 768;
    float* out       = (float*)d_out;
    float* probs_out = out + M * E;            // [12][256][256] f32

    char* ws = (char*)d_ws;
    size_t off = 0;
    auto walloc = [&](size_t bytes) -> char* {
        char* p = ws + off;
        off += (bytes + 255) & ~(size_t)255;
        return p;
    };
    unsigned short* xb   = (unsigned short*)walloc(M * E * 2);
    unsigned short* wqt  = (unsigned short*)walloc(E * E * 2);
    unsigned short* wkt  = (unsigned short*)walloc(E * E * 2);
    unsigned short* wvt  = (unsigned short*)walloc(E * E * 2);
    unsigned short* wot  = (unsigned short*)walloc(E * E * 2);
    unsigned short* qb   = (unsigned short*)walloc(M * E * 2);
    unsigned short* kb_  = (unsigned short*)walloc(M * E * 2);
    unsigned short* vb   = (unsigned short*)walloc(M * E * 2);
    unsigned short* ctxb = (unsigned short*)walloc(M * E * 2);
    unsigned short* pbf  = (unsigned short*)walloc((size_t)12 * 256 * 256 * 2);

    const float scaling = 0.125f / 16.0f;   // Dh^-0.5 / sqrt(R) = 1/128

    // Stage 1: convert inputs to bf16
    long long n4 = (long long)(M * E / 4);
    convert_f32_bf16_kernel<<<(unsigned)((n4 + 255) / 256), 256, 0, stream>>>(x, xb, n4);
    dim3 tb(32, 8);
    transpose_convert_768_kernel<<<dim3(24, 24), tb, 0, stream>>>(Wq, wqt);
    transpose_convert_768_kernel<<<dim3(24, 24), tb, 0, stream>>>(Wk, wkt);
    transpose_convert_768_kernel<<<dim3(24, 24), tb, 0, stream>>>(Wv, wvt);
    transpose_convert_768_kernel<<<dim3(24, 24), tb, 0, stream>>>(Wo, wot);

    // Stage 2: QKV projections (scaling folded into q)
    dim3 gproj((unsigned)(M / GM_BM), (unsigned)(E / GM_BN));   // (512, 12)
    gemm_bf16_kernel<true><<<gproj, 256, 0, stream>>>(xb, wqt, qb,  (int)M, (int)E, (int)E, bq, scaling);
    gemm_bf16_kernel<true><<<gproj, 256, 0, stream>>>(xb, wkt, kb_, (int)M, (int)E, (int)E, bk, 1.0f);
    gemm_bf16_kernel<true><<<gproj, 256, 0, stream>>>(xb, wvt, vb,  (int)M, (int)E, (int)E, bv, 1.0f);

    // Stage 3: scores -> f32 in d_out probs region
    scores_kernel<<<dim3(4, 4, 12), 256, 0, stream>>>(qb, kb_, probs_out);

    // Stage 4: softmax over j; bf16 copy for context GEMM
    softmax_kernel<<<12 * 256, 256, 0, stream>>>(probs_out, pbf);

    // Stage 5: context = probs x V per (r, h)
    context_kernel<<<dim3(4, 12, 256), 256, 0, stream>>>(pbf, vb, ctxb);

    // Stage 6: output projection -> f32 d_out
    gemm_bf16_kernel<false><<<gproj, 256, 0, stream>>>(ctxb, wot, out, (int)M, (int)E, (int)E, bo, 1.0f);
}
